// Complex_Conv1D_6949257085130
// MI455X (gfx1250) — matrix operand order repaired and checked by reading, not compiled, not run on hardware
//
#include <hip/hip_runtime.h>
#include <hip/hip_bf16.h>
#include <stdint.h>

typedef __attribute__((ext_vector_type(16))) _Float16 v16h;
typedef __attribute__((ext_vector_type(8)))  _Float16 v8h;
typedef __attribute__((ext_vector_type(4)))  _Float16 v4h;
typedef __attribute__((ext_vector_type(8)))  float    v8f;
typedef __attribute__((ext_vector_type(4)))  float    v4f;

#define B_    32
#define L_    4096
#define CIN   64
#define KW    9
#define F_    64
#define LOUT  4088          // L - K + 1
#define NCOL  128           // F*2 (real/imag interleaved == output layout)
#define NSTEP 36            // K-steps of 32: 2*KW*CIN / 32
#define TM    64            // output rows per workgroup
#define XROWS 72            // TM + KW - 1 rows of x staged in LDS
#define XELEM (2 * XROWS * CIN)   // 9216 staged f32 elements

// ---------------------------------------------------------------------------
// Kernel 1: repack complex conv weights into WMMA B-fragment layout, split
// into f16 hi + f16 lo parts (hi/lo split recovers ~22 mantissa bits).
// Combined K' x N weight matrix W (K'=1152, N=128):
//   kk = part*576 + k*64 + c ; n = f*2 + p
//   W[kk][n] = p==0 ? (part==0 ?  wr : -wi) : (part==0 ? wi : wr)
// Fragment layout per 16x16 N-tile t of K-step s (B operand, wave32):
//   lane = (n%16) + 16*(krel>=16), halfword h = krel%16
//   stored at  [((s*8 + t)*32 + lane)*16 + h]
// ---------------------------------------------------------------------------
__global__ __launch_bounds__(256) void pack_weights(
    const float* __restrict__ wr, const float* __restrict__ wi,
    _Float16* __restrict__ bhi, _Float16* __restrict__ blo)
{
    int gid = blockIdx.x * 256 + threadIdx.x;      // one thread per (s,t,lane)
    if (gid >= NSTEP * 8 * 32) return;
    int lane = gid & 31;
    int t    = (gid >> 5) & 7;
    int s    = gid >> 8;
    int n    = t * 16 + (lane & 15);
    int f    = n >> 1;
    int p    = n & 1;
    v16h hi, lo;
#pragma unroll
    for (int h = 0; h < 16; ++h) {
        int krel = h + ((lane >> 4) << 4);
        int kk   = s * 32 + krel;
        int part = (kk >= 576) ? 1 : 0;
        int r    = kk - part * 576;
        int k    = r >> 6;
        int c    = r & 63;
        size_t widx = (size_t)((k * 64 + c) * 64 + f);
        float wrv = wr[widx];
        float wiv = wi[widx];
        float val = (p == 0) ? (part == 0 ? wrv : -wiv)
                             : (part == 0 ? wiv :  wrv);
        _Float16 hv = (_Float16)val;
        hi[h] = hv;
        lo[h] = (_Float16)(val - (float)hv);
    }
    *(v16h*)(bhi + (size_t)gid * 16) = hi;
    *(v16h*)(blo + (size_t)gid * 16) = lo;
}

// ---------------------------------------------------------------------------
// Kernel 2: implicit-GEMM complex conv.
// Grid: (L_/TM, B_) ; block = 256 (8 waves).
// Stage 1: async global->LDS copy of the x window (ASYNCcnt path).
// Stage 2: one-time cooperative f32 -> f16 hi/lo split in LDS.
// Stage 3: K-loop = pure feed: ds_load_b128 (A) + global_load_b128 (B, L2)
//          + 12 v_wmma_f32_16x16x32_f16 per step (hi*hi + hi*lo + lo*hi).
// ---------------------------------------------------------------------------
__global__ __launch_bounds__(256) void cconv_wmma(
    const float* __restrict__ xr, const float* __restrict__ xi,
    const _Float16* __restrict__ bhi, const _Float16* __restrict__ blo,
    const float* __restrict__ bias_r, const float* __restrict__ bias_i,
    float* __restrict__ out)
{
    __shared__ __align__(16) float    xs[XELEM];   // 36 KB f32 staging
    __shared__ __align__(16) _Float16 xh[XELEM];   // 18 KB hi
    __shared__ __align__(16) _Float16 xl[XELEM];   // 18 KB lo
    const int tid  = threadIdx.x;
    const int bidx = blockIdx.y;
    const int l0   = blockIdx.x * TM;

    // ---- stage 1: async copy, 2 parts x 72 rows x 64 ch = 2304 x b128 ----
    uint32_t lds0 = (uint32_t)(uintptr_t)(&xs[0]); // low 32 bits = LDS offset
#pragma unroll
    for (int it = 0; it < 9; ++it) {
        int i    = tid + it * 256;                 // 16-byte chunk index
        int e    = i * 4;                          // f32 element index
        int part = (e >= XROWS * CIN) ? 1 : 0;
        int rem  = e - part * (XROWS * CIN);
        int row  = rem >> 6;
        int c    = rem & 63;
        int rowg = l0 + row;
        if (rowg > L_ - 1) rowg = L_ - 1;          // clamp tail reads in-bounds
        const float* src = (part ? xi : xr) +
                           ((size_t)(bidx * L_ + rowg) * CIN + c);
        uint32_t dst = lds0 + (uint32_t)e * 4u;
        asm volatile("global_load_async_to_lds_b128 %0, %1, off"
                     :: "v"(dst), "v"(src) : "memory");
    }
    asm volatile("s_wait_asynccnt 0" ::: "memory");
    __syncthreads();

    // ---- stage 2: split f32 -> f16 hi/lo once (36 elements per thread) ----
#pragma unroll
    for (int it = 0; it < 9; ++it) {
        int e = (tid + it * 256) * 4;
        v4f v = *(const v4f*)(xs + e);
        v4h hi, lo;
#pragma unroll
        for (int j = 0; j < 4; ++j) {
            _Float16 h = (_Float16)v[j];
            hi[j] = h;
            lo[j] = (_Float16)(v[j] - (float)h);
        }
        *(v4h*)(xh + e) = hi;
        *(v4h*)(xl + e) = lo;
    }
    __syncthreads();

    const int w    = tid >> 5;
    const int lane = tid & 31;
    const int mt   = w & 3;                        // M-tile within 64 rows
    const int nh   = w >> 2;                       // N-half (0 or 1)
    const int lh   = lane >> 4;                    // lane half
    const int mrow = mt * 16 + (lane & 15);        // A row within tile block

    v8f acc[4] = {v8f{}, v8f{}, v8f{}, v8f{}};

#pragma unroll 2
    for (int s = 0; s < NSTEP; ++s) {
        int part = (s >= 18) ? 1 : 0;
        int q    = s - part * 18;
        int k    = q >> 1;                         // conv tap
        int c0   = (q & 1) << 5;                   // channel half (0/32)

        // A fragments: 16 contiguous halves per lane, k = 16*lh + s (matches B packing)
        int base = (part * XROWS + mrow + k) * CIN + c0 + lh * 16;
        v8h ah0 = *(const v8h*)(xh + base);
        v8h ah1 = *(const v8h*)(xh + base + 8);
        v8h al0 = *(const v8h*)(xl + base);
        v8h al1 = *(const v8h*)(xl + base + 8);
        v16h ahi = __builtin_shufflevector(ah0, ah1,
                       0,1,2,3,4,5,6,7,8,9,10,11,12,13,14,15);
        v16h alo = __builtin_shufflevector(al0, al1,
                       0,1,2,3,4,5,6,7,8,9,10,11,12,13,14,15);

        const _Float16* bp = bhi + (((size_t)s * 8 + nh * 4) * 32 + lane) * 16;
        const _Float16* lp = blo + (((size_t)s * 8 + nh * 4) * 32 + lane) * 16;
#pragma unroll
        for (int tt = 0; tt < 4; ++tt) {
            v16h bh = *(const v16h*)(bp + (size_t)tt * 512);
            v16h bl = *(const v16h*)(lp + (size_t)tt * 512);
            v8f a = acc[tt];
            a = __builtin_amdgcn_wmma_f32_16x16x32_f16(false, ahi, false, bh,
                                                       (short)0, a, false, false);
            a = __builtin_amdgcn_wmma_f32_16x16x32_f16(false, ahi, false, bl,
                                                       (short)0, a, false, false);
            a = __builtin_amdgcn_wmma_f32_16x16x32_f16(false, alo, false, bh,
                                                       (short)0, a, false, false);
            acc[tt] = a;
        }
    }

    // ---- store D tiles (+bias); D layout: VGPR r -> M=r+8*lh, n=lane%16 ----
#pragma unroll
    for (int tt = 0; tt < 4; ++tt) {
        int col = nh * 64 + tt * 16 + (lane & 15);
        float bias = (col & 1) ? bias_i[col >> 1] : bias_r[col >> 1];
#pragma unroll
        for (int r = 0; r < 8; ++r) {
            int l = l0 + mt * 16 + r + 8 * lh;
            if (l < LOUT) {
                out[(size_t)(bidx * LOUT + l) * NCOL + col] = acc[tt][r] + bias;
            }
        }
    }
}

extern "C" void kernel_launch(void* const* d_in, const int* in_sizes, int n_in,
                              void* d_out, int out_size, void* d_ws, size_t ws_size,
                              hipStream_t stream) {
    const float* xr = (const float*)d_in[0];
    const float* xi = (const float*)d_in[1];
    const float* wr = (const float*)d_in[2];
    const float* wi = (const float*)d_in[3];
    const float* br = (const float*)d_in[4];
    const float* bi = (const float*)d_in[5];
    float* out = (float*)d_out;

    _Float16* bhi = (_Float16*)d_ws;                       // 288 KB
    _Float16* blo = bhi + (size_t)NSTEP * 8 * 32 * 16;     // +288 KB

    pack_weights<<<NSTEP, 256, 0, stream>>>(wr, wi, bhi, blo);

    dim3 grid(L_ / TM, B_);                                // 64 x 32
    cconv_wmma<<<grid, dim3(256), 0, stream>>>(xr, xi, bhi, blo, br, bi, out);
}
